// BasicGRU4rec_36318243455036
// MI455X (gfx1250) — compile-verified, matrix operand order checked by reading, not run on hardware
//
#include <hip/hip_runtime.h>
#include <hip/hip_bf16.h>

typedef __attribute__((ext_vector_type(16))) __bf16 v16bf;
typedef __attribute__((ext_vector_type(8)))  __bf16 v8bf;
typedef __attribute__((ext_vector_type(8)))  float  v8f;

#define GB 1024
#define GT 200
#define GH 128
#define G3H 384
#define GR (GB*GT)        // 204800 flat rows
#define GMT (GR/16)       // 12800 M-tiles

static __device__ __forceinline__ v16bf load_frag_bf(const __bf16* p) {
  // two contiguous 8x-bf16 (16B) chunks: k in [0,8) and [16,24) relative to p
  union { v16bf v; v8bf h[2]; } u;
  u.h[0] = *reinterpret_cast<const v8bf*>(p);
  u.h[1] = *reinterpret_cast<const v8bf*>(p + 16);
  return u.v;
}

static __device__ __forceinline__ v16bf load_frag_f32(const float* p) {
  union { v16bf v; __bf16 e[16]; } u;
#pragma unroll
  for (int i = 0; i < 8; ++i) {
    u.e[i]     = (__bf16)p[i];
    u.e[8 + i] = (__bf16)p[16 + i];
  }
  return u.v;
}

static __device__ __forceinline__ v8f wmma_bf16(v16bf a, v16bf b, v8f c) {
  return __builtin_amdgcn_wmma_f32_16x16x32_bf16(false, a, false, b, (short)0, c, false, false);
}

// ---------------- K0: fp32 -> bf16 weight convert ----------------
__global__ void cvt_bf16_kernel(const float* __restrict__ src, __bf16* __restrict__ dst, int n) {
  int i = blockIdx.x * blockDim.x + threadIdx.x;
  if (i < n) dst[i] = (__bf16)src[i];
}

// ---------------- K1: embedding gathers ----------------
__global__ void gather_kernel(const int* __restrict__ ilist, const int* __restrict__ nlist,
                              const float* __restrict__ emb,
                              float* __restrict__ out2, float* __restrict__ out3,
                              float* __restrict__ out4) {
  int r = blockIdx.x;            // flat (b,t)
  int j = threadIdx.x;           // 0..127
  int b = r / GT, t = r % GT;
  float e = emb[(size_t)ilist[r] * GH + j];
  out4[(size_t)r * (2 * GH) + j] = e;
  if (t >= 1) out2[((size_t)b * (GT - 1) + (t - 1)) * GH + j] = e;
  float en = emb[(size_t)nlist[r] * GH + j];
  if (t < GT - 1) out3[((size_t)b * (GT - 1) + t) * GH + j] = en;
}

// ---------------- K2/K4: gi = A(128) x W^T(384x128) + bias, all B*T rows ----------------
__global__ __launch_bounds__(256) void gemm_gi_kernel(const float* __restrict__ Af32,
                                                      const __bf16* __restrict__ Abf,
                                                      int astride,
                                                      const __bf16* __restrict__ W,   // (384,128) row-major
                                                      const float* __restrict__ bias, // (384)
                                                      float* __restrict__ out) {      // (B*T,384)
  int wave = threadIdx.x >> 5;
  int lane = threadIdx.x & 31;
  int tile = blockIdx.x * 8 + wave;
  if (tile >= GMT) return;
  int r0 = tile * 16;
  int mlo = lane & 15;
  int khalf = (lane >> 4) & 1;
  int koff = khalf * 8;

  v16bf a[4];
#pragma unroll
  for (int kt = 0; kt < 4; ++kt) {
    int base = kt * 32 + koff;
    if (Af32) a[kt] = load_frag_f32(Af32 + (size_t)(r0 + mlo) * astride + base);
    else      a[kt] = load_frag_bf (Abf  + (size_t)(r0 + mlo) * astride + base);
  }
#pragma unroll
  for (int nt = 0; nt < 24; ++nt) {
    int n = nt * 16 + mlo;
    float bv = bias[n];
    v8f c = {};                       // first WMMA uses inline-0 SRC2
#pragma unroll
    for (int kt = 0; kt < 4; ++kt) {
      v16bf bf = load_frag_bf(W + (size_t)n * GH + kt * 32 + koff);
      c = wmma_bf16(a[kt], bf, c);
    }
#pragma unroll
    for (int s = 0; s < 8; ++s) {
      int row = r0 + s + khalf * 8;
      out[(size_t)row * G3H + n] = c[s] + bv;   // bias folded at store
    }
  }
}

// ---------------- K3/K5: sequential GRU layer ----------------
// 64 blocks x 128 threads; block owns 16 batch rows for all T steps.
#define WSTR 136   // padded bf16 row stride for Whh in LDS (272B rows, 16B-aligned chunks)
#define GSTR 392   // padded f32 row stride for gh buffer
#define HSTRF 132
#define HSTRB 136
__global__ __launch_bounds__(128) void gru_seq_kernel(const float* __restrict__ gi,   // (B*T,384), includes bih
                                                      const __bf16* __restrict__ Wb,  // Whh bf16 (384,128)
                                                      const float* __restrict__ bhh,  // (384)
                                                      const float* __restrict__ mask, // (B,T) flat == row index
                                                      __bf16* __restrict__ h_all) {   // (B*T,128) unmasked h
  __shared__ __bf16 sW[G3H * WSTR];
  __shared__ float  sGH[16 * GSTR];
  __shared__ float  sHf[16 * HSTRF];
  __shared__ __bf16 sHb[16 * HSTRB];

  int tid = threadIdx.x;
  int wave = tid >> 5, lane = tid & 31;
  int mlo = lane & 15, khalf = (lane >> 4) & 1, koff = khalf * 8;
  int bb = blockIdx.x * 16;

  // Stage Whh into LDS via async memory->LDS path (16B chunks, padded rows).
  {
    unsigned lds_base = (unsigned)(uintptr_t)(void*)&sW[0];
    unsigned long long wbase = (unsigned long long)(uintptr_t)Wb;
    for (int i = tid; i < G3H * 16; i += 128) {     // 384 rows x 16 chunks of 16B
      unsigned rr = (unsigned)(i >> 4), c = (unsigned)(i & 15);
      unsigned voff = rr * 256u + c * 16u;          // source byte offset (stride 256B)
      unsigned dst  = lds_base + rr * 272u + c * 16u; // dest byte offset (stride 272B)
      asm volatile("global_load_async_to_lds_b128 %0, %1, %2"
                   :: "v"(dst), "v"(voff), "s"(wbase) : "memory");
    }
    asm volatile("s_wait_asynccnt 0" ::: "memory");
  }
  for (int i = tid; i < 16 * HSTRF; i += 128) sHf[i] = 0.0f;
  for (int i = tid; i < 16 * HSTRB; i += 128) sHb[i] = (__bf16)0.0f;
  __syncthreads();

  // hoisted per-wave constants: bhh for this wave's 6 N-tiles
  float bvq[6];
#pragma unroll
  for (int q = 0; q < 6; ++q) bvq[q] = bhh[(wave * 6 + q) * 16 + mlo];

  for (int t = 0; t < GT; ++t) {
    // 1) issue this step's gi + mask loads early; they ride out the WMMA phase
    float gr0[16], gz0[16], gn0[16], keepv[4];
#pragma unroll
    for (int mi = 0; mi < 4; ++mi) {
      int m = wave * 4 + mi;
      int r = (bb + m) * GT + t;
      const float* girow = gi + (size_t)r * G3H;
      keepv[mi] = mask[r];
#pragma unroll
      for (int jj = 0; jj < 4; ++jj) {
        int j = lane + jj * 32;
        gr0[mi * 4 + jj] = girow[j];
        gz0[mi * 4 + jj] = girow[128 + j];
        gn0[mi * 4 + jj] = girow[256 + j];
      }
    }

    // 2) gh = h @ Whh^T  (16x384 tile via WMMA, 6 N-tiles per wave)
    v16bf a[4];
#pragma unroll
    for (int kt = 0; kt < 4; ++kt)
      a[kt] = load_frag_bf(&sHb[mlo * HSTRB + kt * 32 + koff]);
#pragma unroll
    for (int q = 0; q < 6; ++q) {
      int nt = wave * 6 + q;
      int n = nt * 16 + mlo;
      v8f c = {};
#pragma unroll
      for (int kt = 0; kt < 4; ++kt) {
        v16bf bf = load_frag_bf(&sW[n * WSTR + kt * 32 + koff]);
        c = wmma_bf16(a[kt], bf, c);
      }
#pragma unroll
      for (int s = 0; s < 8; ++s)
        sGH[(s + khalf * 8) * GSTR + n] = c[s] + bvq[q];   // + bhh
    }
    __syncthreads();

    // 3) gate math from registers + LDS
#pragma unroll
    for (int mi = 0; mi < 4; ++mi) {
      int m = wave * 4 + mi;
      int r = (bb + m) * GT + t;
      float keep = (keepv[mi] != 0.0f) ? 1.0f : 0.0f;
#pragma unroll
      for (int jj = 0; jj < 4; ++jj) {
        int j = lane + jj * 32;
        float hr = sGH[m * GSTR + j], hz = sGH[m * GSTR + 128 + j], hn = sGH[m * GSTR + 256 + j];
        float hprev = sHf[m * HSTRF + j];
        float rg = 1.0f / (1.0f + __expf(-(gr0[mi * 4 + jj] + hr)));
        float zg = 1.0f / (1.0f + __expf(-(gz0[mi * 4 + jj] + hz)));
        float ng = tanhf(gn0[mi * 4 + jj] + rg * hn);
        float hnew = (1.0f - zg) * ng + zg * hprev;
        h_all[(size_t)r * GH + j] = (__bf16)hnew;   // unmasked, fed forward
        float hm = hnew * keep;                      // masked, carried
        sHf[m * HSTRF + j] = hm;
        sHb[m * HSTRB + j] = (__bf16)hm;
      }
    }
    __syncthreads();
  }
}

// ---------------- K6: out = tanh(h1 @ h2o_w^T + b) -> outputs 1 and 4[:,:,128:256] ----------------
__global__ __launch_bounds__(256) void proj_kernel(const __bf16* __restrict__ h1,   // (B*T,128)
                                                   const __bf16* __restrict__ W,    // (128,128)
                                                   const float* __restrict__ bias,  // (128)
                                                   float* __restrict__ out1,        // (B,T-1,128)
                                                   float* __restrict__ out4) {      // (B,T,256)
  int wave = threadIdx.x >> 5;
  int lane = threadIdx.x & 31;
  int tile = blockIdx.x * 8 + wave;
  if (tile >= GMT) return;
  int r0 = tile * 16;
  int mlo = lane & 15, khalf = (lane >> 4) & 1, koff = khalf * 8;

  v16bf a[4];
#pragma unroll
  for (int kt = 0; kt < 4; ++kt)
    a[kt] = load_frag_bf(h1 + (size_t)(r0 + mlo) * GH + kt * 32 + koff);
#pragma unroll
  for (int nt = 0; nt < 8; ++nt) {
    int n = nt * 16 + mlo;
    float bv = bias[n];
    v8f c = {};
#pragma unroll
    for (int kt = 0; kt < 4; ++kt) {
      v16bf bf = load_frag_bf(W + (size_t)n * GH + kt * 32 + koff);
      c = wmma_bf16(a[kt], bf, c);
    }
#pragma unroll
    for (int s = 0; s < 8; ++s) {
      int row = r0 + s + khalf * 8;
      float v = tanhf(c[s] + bv);
      int b = row / GT, t = row % GT;
      out4[(size_t)row * (2 * GH) + GH + n] = v;
      if (t < GT - 1) out1[((size_t)b * (GT - 1) + t) * GH + n] = v;
    }
  }
}

extern "C" void kernel_launch(void* const* d_in, const int* in_sizes, int n_in,
                              void* d_out, int out_size, void* d_ws, size_t ws_size,
                              hipStream_t stream) {
  (void)in_sizes; (void)n_in; (void)out_size; (void)ws_size;
  const int*   ilist = (const int*)d_in[0];
  const float* imask = (const float*)d_in[1];
  const int*   nlist = (const int*)d_in[2];
  const float* emb   = (const float*)d_in[3];
  const float* Wih   = (const float*)d_in[4];   // (2,384,128)
  const float* Whh   = (const float*)d_in[5];
  const float* bih   = (const float*)d_in[6];   // (2,384)
  const float* bhh   = (const float*)d_in[7];
  const float* h2ow  = (const float*)d_in[8];   // (128,128)
  const float* h2ob  = (const float*)d_in[9];

  float* out  = (float*)d_out;
  float* out1 = out;                                        // (B,T-1,H)
  float* out2 = out + (size_t)GB * (GT - 1) * GH;           // pos targets
  float* out3 = out2 + (size_t)GB * (GT - 1) * GH;          // neg targets
  float* out4 = out3 + (size_t)GB * (GT - 1) * GH;          // (B,T,2H)

  // workspace layout
  char* ws = (char*)d_ws;
  __bf16* wih0b = (__bf16*)(ws);                   // 49152
  __bf16* whh0b = wih0b + 49152;
  __bf16* wih1b = whh0b + 49152;
  __bf16* whh1b = wih1b + 49152;
  __bf16* h2owb = whh1b + 49152;                   // 16384
  float*  gi    = (float*)(ws + 425984);           // (B*T,384) fp32, reused by both layers
  __bf16* h0all = (__bf16*)(ws + 425984 + (size_t)GR * G3H * 4);
  __bf16* h1all = h0all + (size_t)GR * GH;

  // K0: convert weights to bf16
  cvt_bf16_kernel<<<(49152 + 255) / 256, 256, 0, stream>>>(Wih,          wih0b, 49152);
  cvt_bf16_kernel<<<(49152 + 255) / 256, 256, 0, stream>>>(Whh,          whh0b, 49152);
  cvt_bf16_kernel<<<(49152 + 255) / 256, 256, 0, stream>>>(Wih + 49152,  wih1b, 49152);
  cvt_bf16_kernel<<<(49152 + 255) / 256, 256, 0, stream>>>(Whh + 49152,  whh1b, 49152);
  cvt_bf16_kernel<<<(16384 + 255) / 256, 256, 0, stream>>>(h2ow,         h2owb, 16384);

  // K1: gathers (outputs 2, 3, and out4 first half = input embeddings)
  gather_kernel<<<GR, 128, 0, stream>>>(ilist, nlist, emb, out2, out3, out4);

  // K2: gi0 for all (b,t) — big parallel WMMA GEMM (A = fp32 embeddings in out4, stride 256)
  gemm_gi_kernel<<<GMT / 8, 256, 0, stream>>>(out4, nullptr, 2 * GH, wih0b, bih, gi);

  // K3: layer-0 recurrence
  gru_seq_kernel<<<GB / 16, 128, 0, stream>>>(gi, whh0b, bhh, imask, h0all);

  // K4: gi1 for all (b,t) from h0 (bf16, stride 128)
  gemm_gi_kernel<<<GMT / 8, 256, 0, stream>>>(nullptr, h0all, GH, wih1b, bih + G3H, gi);

  // K5: layer-1 recurrence
  gru_seq_kernel<<<GB / 16, 128, 0, stream>>>(gi, whh1b, bhh + G3H, imask, h1all);

  // K6: output projection + tanh -> outputs 1 and 4[:,:,128:]
  proj_kernel<<<GMT / 8, 256, 0, stream>>>(h1all, h2owb, h2ob, out1, out4);
}
